// TokenSetRouter_55310588838263
// MI455X (gfx1250) — matrix operand order repaired and checked by reading, not compiled, not run on hardware
//
#include <hip/hip_runtime.h>

// ---------- types ----------
typedef __attribute__((ext_vector_type(16))) __bf16 v16bf;
typedef __attribute__((ext_vector_type(8)))  float  v8f;

union FragBF { uint4 q[2]; v16bf v; };

__device__ __forceinline__ unsigned short f2bf(float f) {
    union { float f; unsigned u; } x; x.f = f;
    unsigned r = x.u + 0x7FFFu + ((x.u >> 16) & 1u);   // round-to-nearest-even
    return (unsigned short)(r >> 16);
}

// CDNA5 async global->LDS path (ASYNCcnt), guarded so absence can't break compile.
#if __has_builtin(__builtin_amdgcn_global_load_async_to_lds_b128) && \
    __has_builtin(__builtin_amdgcn_s_wait_asynccnt)
#define HAS_ASYNC_LDS 1
#else
#define HAS_ASYNC_LDS 0
#endif

#if HAS_ASYNC_LDS
// Builtin expects: (int4* gaddr, int4 AS(3)* lds, imm offset, imm cpol)
// (diagnostic showed param0 = generic pointer to __vector_size__(16) int, non-const)
typedef int gvec4i __attribute__((vector_size(16)));
typedef __attribute__((address_space(3))) gvec4i* lds_v4i_ptr;
__device__ __forceinline__ void async_copy16(const void* g, void* l) {
    __builtin_amdgcn_global_load_async_to_lds_b128((gvec4i*)g, (lds_v4i_ptr)l, 0, 0);
}
#endif

// ---------- fp32 -> bf16 conversion (vectorized, n multiple of 4) ----------
__global__ __launch_bounds__(256)
void cvt_f32_bf16(const float* __restrict__ in, unsigned short* __restrict__ out, int n4) {
    int i = blockIdx.x * 256 + threadIdx.x;
    if (i < n4) {
        float4 f = ((const float4*)in)[i];
        ushort4 o;
        o.x = f2bf(f.x); o.y = f2bf(f.y); o.z = f2bf(f.z); o.w = f2bf(f.w);
        ((ushort4*)out)[i] = o;
    }
}

// ---------- bf16 NT GEMM: C[M,N] = A[M,K] * B[N,K]^T (+bias) ----------
// MODE 0: f32 out + bias;  MODE 1: bf16 out + bias;  MODE 2: f32 out, no bias
// Block tile 128x128, BK=64, 256 threads = 8 waves (2x4), wave tile 64x32.
// Double-buffered LDS filled by GLOBAL_LOAD_ASYNC_TO_LDS_B128 (ASYNCcnt).
// All of M,N multiples of 128 and K multiple of 64 in this pipeline.
template <int MODE>
__global__ __launch_bounds__(256)
void gemm_bf16_nt(const unsigned short* __restrict__ A,
                  const unsigned short* __restrict__ B,
                  const float* __restrict__ bias,
                  void* __restrict__ Cout,
                  int M, int N, int K,
                  long sA, long sB, long sC)
{
    constexpr int LDT = 72;                    // 64 + 8 pad: conflict-free b128 reads
    __shared__ unsigned short As[2][128 * LDT];
    __shared__ unsigned short Bs[2][128 * LDT];

    const int tid  = threadIdx.x;
    const int lane = tid & 31;
    const int wave = tid >> 5;
    const int wm   = wave & 1;                 // 2 waves along M
    const int wn   = wave >> 1;                // 4 waves along N
    const int bm   = blockIdx.y * 128;
    const int bn   = blockIdx.x * 128;
    const int r0   = lane & 15;
    const int hi   = lane >> 4;

    A += (long)blockIdx.z * sA;
    B += (long)blockIdx.z * sB;

    v8f acc[4][2];
    #pragma unroll
    for (int mi = 0; mi < 4; ++mi)
        #pragma unroll
        for (int ni = 0; ni < 2; ++ni)
            #pragma unroll
            for (int e = 0; e < 8; ++e) acc[mi][ni][e] = 0.0f;

    // This thread's 4 fill chunks (16B each): tile is 1024 x 16B per matrix.
    // chunk c: row = c>>3 (0..127), col = (c&7)*8 bf16 elements.
    auto issue_fill = [&](int buf, int kt) {
        #pragma unroll
        for (int i = 0; i < 4; ++i) {
            int c   = tid + i * 256;
            int row = c >> 3;
            int col = (c & 7) * 8;
            const unsigned short* ga = &A[(long)(bm + row) * K + kt + col];
            const unsigned short* gb = &B[(long)(bn + row) * K + kt + col];
            unsigned short* la = &As[buf][row * LDT + col];
            unsigned short* lb = &Bs[buf][row * LDT + col];
#if HAS_ASYNC_LDS
            async_copy16(ga, la);
            async_copy16(gb, lb);
#else
            *(uint4*)la = *(const uint4*)ga;
            *(uint4*)lb = *(const uint4*)gb;
#endif
        }
    };

    issue_fill(0, 0);

    int buf = 0;
    for (int kt = 0; kt < K; kt += 64, buf ^= 1) {
#if HAS_ASYNC_LDS
        __builtin_amdgcn_s_wait_asynccnt(0);   // own fills for `buf` have landed
#endif
        __syncthreads();                       // all waves' fills visible; spare buf free
        if (kt + 64 < K) issue_fill(buf ^ 1, kt + 64);

        const unsigned short* Asb = As[buf];
        const unsigned short* Bsb = Bs[buf];
        #pragma unroll
        for (int ks = 0; ks < 2; ++ks) {       // two K=32 WMMA steps per tile
            FragBF a[4], b[2];
            #pragma unroll
            for (int mi = 0; mi < 4; ++mi) {
                const unsigned short* p =
                    &Asb[(wm * 64 + mi * 16 + r0) * LDT + ks * 32 + hi * 8];
                a[mi].q[0] = *(const uint4*)(p);        // K = kb .. kb+7
                a[mi].q[1] = *(const uint4*)(p + 16);   // K = kb+16 .. kb+23
            }
            #pragma unroll
            for (int ni = 0; ni < 2; ++ni) {
                const unsigned short* p =
                    &Bsb[(wn * 32 + ni * 16 + r0) * LDT + ks * 32 + hi * 8];
                b[ni].q[0] = *(const uint4*)(p);        // K = kb .. kb+7
                b[ni].q[1] = *(const uint4*)(p + 16);   // K = kb+16 .. kb+23
            }
            #pragma unroll
            for (int mi = 0; mi < 4; ++mi)
                #pragma unroll
                for (int ni = 0; ni < 2; ++ni)
                    acc[mi][ni] = __builtin_amdgcn_wmma_f32_16x16x32_bf16(
                        false, a[mi].v, false, b[ni].v,
                        (short)0, acc[mi][ni], false, false);
        }
    }

    __syncthreads();                           // last tile fully consumed

    // ---- epilogue ----
    #pragma unroll
    for (int mi = 0; mi < 4; ++mi) {
        #pragma unroll
        for (int ni = 0; ni < 2; ++ni) {
            int gn = bn + wn * 32 + ni * 16 + r0;
            float bv = (MODE == 2) ? 0.0f : bias[gn];
            #pragma unroll
            for (int e = 0; e < 8; ++e) {
                int gm = bm + wm * 64 + mi * 16 + hi * 8 + e;
                long coff = (long)blockIdx.z * sC + (long)gm * N + gn;
                float v = acc[mi][ni][e] + bv;
                if (MODE == 1) ((unsigned short*)Cout)[coff] = f2bf(v);
                else           ((float*)Cout)[coff] = v;
            }
        }
    }
}

// ---------- top-8 + softmax, one wave32 per (b,l) token, S=256 ----------
__global__ __launch_bounds__(256)
void topk_softmax(const float* __restrict__ logits,
                  const unsigned char* __restrict__ mask,
                  float* __restrict__ probs, int* __restrict__ outidx)
{
    const int wave  = threadIdx.x >> 5;
    const int lane  = threadIdx.x & 31;
    const int token = blockIdx.x * 8 + wave;
    const int b     = token >> 12;             // L = 4096
    const float NEGINF = -__builtin_inff();

    const float* row = logits + (long)token * 256;
    float v[8]; int si[8];
    #pragma unroll
    for (int j = 0; j < 8; ++j) {
        int s = j * 32 + lane;
        float x = row[s];
        if (!mask[b * 256 + s]) x = NEGINF;
        v[j] = x; si[j] = s;
    }

    float topv[8]; int topi[8];
    #pragma unroll
    for (int k = 0; k < 8; ++k) {
        float bv = v[0]; int bi = si[0];
        #pragma unroll
        for (int j = 1; j < 8; ++j)
            if (v[j] > bv) { bv = v[j]; bi = si[j]; }
        #pragma unroll
        for (int off = 16; off > 0; off >>= 1) {         // wave32 butterfly
            float ov = __shfl_xor(bv, off, 32);
            int   oi = __shfl_xor(bi, off, 32);
            if (ov > bv || (ov == bv && oi < bi)) { bv = ov; bi = oi; }
        }
        topv[k] = bv; topi[k] = bi;
        if ((bi & 31) == lane) v[bi >> 5] = NEGINF;      // retire winner
    }

    float m = topv[0];
    float pe[8], sum = 0.0f;
    #pragma unroll
    for (int k = 0; k < 8; ++k) { pe[k] = __expf(topv[k] - m); sum += pe[k]; }
    float inv = 1.0f / sum;
    if (lane < 8) {
        probs [token * 8 + lane] = pe[lane] * inv;
        outidx[token * 8 + lane] = topi[lane];
    }
}

// ---------- sparse mix: mix[t,:] = sum_k p[k] * Z[b, idx[k], :] ; bf16 out ----------
__global__ __launch_bounds__(256)
void mix_gather(const float* __restrict__ Z,       // (B,S,1024) f32
                const float* __restrict__ probs,
                const int*   __restrict__ idx,
                unsigned short* __restrict__ mixout)
{
    const int token = blockIdx.x;
    const int b     = token >> 12;             // L = 4096
    const int d     = threadIdx.x * 4;
    float4 acc = make_float4(0.f, 0.f, 0.f, 0.f);
    #pragma unroll
    for (int k = 0; k < 8; ++k) {
        float p = probs[token * 8 + k];
        int   s = idx  [token * 8 + k];
        float4 z = *(const float4*)(Z + ((long)(b * 256 + s)) * 1024 + d);
        acc.x += p * z.x; acc.y += p * z.y; acc.z += p * z.z; acc.w += p * z.w;
    }
    ushort4 o;
    o.x = f2bf(acc.x); o.y = f2bf(acc.y); o.z = f2bf(acc.z); o.w = f2bf(acc.w);
    *(ushort4*)(mixout + (long)token * 1024 + d) = o;
}

// ---------- launcher ----------
extern "C" void kernel_launch(void* const* d_in, const int* in_sizes, int n_in,
                              void* d_out, int out_size, void* d_ws, size_t ws_size,
                              hipStream_t stream) {
    constexpr int B = 4, L = 4096, D = 1024, S = 256;
    constexpr long BL = (long)B * L;           // 16384 tokens

    const float*         tok   = (const float*)d_in[0];
    const float*         Z     = (const float*)d_in[1];
    const float*         desc  = (const float*)d_in[2];
    const unsigned char* mask  = (const unsigned char*)d_in[3];
    const float*         Wg_w  = (const float*)d_in[4];
    const float*         Wg_b  = (const float*)d_in[5];
    const float*         Wd_w  = (const float*)d_in[6];
    const float*         Wd_b  = (const float*)d_in[7];
    const float*         ow    = (const float*)d_in[8];
    const float*         ob    = (const float*)d_in[9];
    float*               out   = (float*)d_out;

    // workspace carve-up (256B aligned)
    char* w = (char*)d_ws;
    size_t off = 0;
    auto alloc = [&](size_t bytes) {
        void* p = w + off;
        off = (off + bytes + 255) & ~(size_t)255;
        return p;
    };
    unsigned short* tokBF   = (unsigned short*)alloc(BL * D * 2);
    unsigned short* wgBF    = (unsigned short*)alloc((size_t)D * D * 2);
    unsigned short* wdBF    = (unsigned short*)alloc((size_t)D * D * 2);
    unsigned short* owBF    = (unsigned short*)alloc((size_t)D * D * 2);
    unsigned short* descBF  = (unsigned short*)alloc((size_t)B * S * D * 2);
    unsigned short* tprojBF = (unsigned short*)alloc(BL * D * 2);
    unsigned short* dprojBF = (unsigned short*)alloc((size_t)B * S * D * 2);
    float*          logits  = (float*)alloc(BL * S * 4);
    float*          probs   = (float*)alloc(BL * 8 * 4);
    int*            tidx    = (int*)alloc(BL * 8 * 4);
    unsigned short* mixBF   = (unsigned short*)alloc(BL * D * 2);
    (void)ws_size; (void)n_in; (void)in_sizes; (void)out_size;

    // 1) fp32 -> bf16 conversions
    auto cvt = [&](const float* src, unsigned short* dst, long n) {
        int n4 = (int)(n / 4);
        hipLaunchKernelGGL(cvt_f32_bf16, dim3((n4 + 255) / 256), dim3(256), 0, stream,
                           src, dst, n4);
    };
    cvt(tok,  tokBF,  BL * D);
    cvt(Wg_w, wgBF,   (long)D * D);
    cvt(Wd_w, wdBF,   (long)D * D);
    cvt(ow,   owBF,   (long)D * D);
    cvt(desc, descBF, (long)B * S * D);

    // 2) Tproj = tok @ Wg^T + Wg_b  -> bf16   (16384x1024x1024)
    hipLaunchKernelGGL((gemm_bf16_nt<1>), dim3(D / 128, BL / 128, 1), dim3(256), 0, stream,
                       tokBF, wgBF, Wg_b, tprojBF, (int)BL, D, D, 0L, 0L, 0L);

    // 3) Dproj = desc @ Wd^T + Wd_b -> bf16   (1024x1024x1024)
    hipLaunchKernelGGL((gemm_bf16_nt<1>), dim3(D / 128, (B * S) / 128, 1), dim3(256), 0, stream,
                       descBF, wdBF, Wd_b, dprojBF, B * S, D, D, 0L, 0L, 0L);

    // 4) logits[b] = Tproj[b] @ Dproj[b]^T -> f32  (batched 4096x256x1024)
    hipLaunchKernelGGL((gemm_bf16_nt<2>), dim3(S / 128, L / 128, B), dim3(256), 0, stream,
                       tprojBF, dprojBF, (const float*)nullptr, logits,
                       L, S, D, (long)L * D, (long)S * D, (long)L * S);

    // 5) top-8 + softmax (wave per token)
    hipLaunchKernelGGL(topk_softmax, dim3(BL / 8), dim3(256), 0, stream,
                       logits, mask, probs, tidx);

    // 6) sparse mix -> bf16
    hipLaunchKernelGGL(mix_gather, dim3(BL), dim3(256), 0, stream,
                       Z, probs, tidx, mixBF);

    // 7) out = mix @ out_w^T + out_b -> f32 (d_out)
    hipLaunchKernelGGL((gemm_bf16_nt<0>), dim3(D / 128, BL / 128, 1), dim3(256), 0, stream,
                       mixBF, owBF, ob, out, (int)BL, D, D, 0L, 0L, 0L);
}
